// DownLayer_56418690400662
// MI455X (gfx1250) — matrix operand order repaired and checked by reading, not compiled, not run on hardware
//
#include <hip/hip_runtime.h>
#include <math.h>

#define NTOK 3136
#define CIN  64
#define COUT 128
#define HIMG 56
#define WIMG 56
#define H2   28
#define W2   28
#define NPIX 784
#define NS   784
#define KCONV 576
#define NPAD 4096
#define LN_EPS 1e-5f

typedef __attribute__((ext_vector_type(16))) __bf16 v16bf;
typedef __attribute__((ext_vector_type(8)))  float  v8f;

union BF16x16 { v16bf v; float4 f4[2]; };

__device__ __forceinline__ v8f wmma_bf16(v16bf a, v16bf b, v8f c) {
  // D = A(16x32 bf16) * B(32x16 bf16) + C(16x16 f32)
  return __builtin_amdgcn_wmma_f32_16x16x32_bf16(false, a, false, b, (short)0, c, false, false);
}

__device__ __forceinline__ v8f zero8() {
  v8f a;
#pragma unroll
  for (int q = 0; q < 8; ++q) a[q] = 0.0f;
  return a;
}

// A-matrix fragment (16-bit, 16x32): lane holds row m=lane%16.
// hi = lane/16 : K = k0 + 16*hi + {0..15}
__device__ __forceinline__ v16bf load_afrag(const __bf16* row, int k0, int hi) {
  BF16x16 u;
  u.f4[0] = *(const float4*)(row + k0 + hi * 16);
  u.f4[1] = *(const float4*)(row + k0 + hi * 16 + 8);
  return u.v;
}
// B-matrix fragment (32x16): lane holds column n=lane%16, K = k0 + 16*hi + {0..15}
__device__ __forceinline__ v16bf load_bfrag(const __bf16* row, int k0, int hi) {
  BF16x16 u;
  const __bf16* p = row + k0 + hi * 16;
  u.f4[0] = *(const float4*)(p);
  u.f4[1] = *(const float4*)(p + 8);
  return u.v;
}
__device__ __forceinline__ v16bf zero_frag() {
  BF16x16 u;
  float4 z; z.x = z.y = z.z = z.w = 0.0f;
  u.f4[0] = z; u.f4[1] = z;
  return u.v;
}

__device__ __forceinline__ void ins3(float v, float& t0, float& t1, float& t2) {
  if (v < t2) {
    if (v < t1) { t2 = t1; if (v < t0) { t1 = t0; t0 = v; } else t1 = v; }
    else t2 = v;
  }
}

// ---------------------------------------------------------------- init
__global__ void k_init(float* all_w, unsigned* d2max, unsigned* aggmax,
                       float* xdown, int Bv) {
  int tid = blockIdx.x * blockDim.x + threadIdx.x;
  int nth = gridDim.x * blockDim.x;
  for (int i = tid; i < Bv * NS; i += nth) all_w[i] = 1e-6f;
  for (int i = tid; i < Bv; i += nth) { d2max[i] = 0u; aggmax[i] = 0u; }
  for (int i = tid; i < Bv * NS * COUT; i += nth) xdown[i] = 0.0f;
}

// ---------------------------------------------------------------- pack to bf16
__global__ void k_pack(const float* x, const float* conv_w, const float* skip_w,
                       __bf16* x_bf, __bf16* convA, __bf16* skipA, int Bv) {
  int tid = blockIdx.x * blockDim.x + threadIdx.x;
  int nth = gridDim.x * blockDim.x;
  for (int i = tid; i < Bv * NTOK * CIN; i += nth) x_bf[i] = (__bf16)x[i];
  // convA[o][k], k = (ky*3+kx)*64 + c  <-  conv_w[o][c][ky][kx]
  for (int i = tid; i < COUT * KCONV; i += nth) {
    int o = i / KCONV, k = i % KCONV;
    int tap = k / CIN, c = k % CIN;
    int ky = tap / 3, kx = tap % 3;
    convA[i] = (__bf16)conv_w[((o * CIN + c) * 3 + ky) * 3 + kx];
  }
  for (int i = tid; i < COUT * CIN; i += nth) skipA[i] = (__bf16)skip_w[i];
}

// ---------------------------------------------------------------- conv 3x3 s2 as implicit WMMA GEMM
// grid (49, B), block 256 (8 waves). wave w -> out channels [16w,16w+16)
__global__ __launch_bounds__(256) void k_conv(const __bf16* x_bf, const __bf16* convA,
                                              const float* conv_b, float* conv_out, int Bv) {
  int b = blockIdx.y;
  int tid = threadIdx.x, w = tid >> 5, lane = tid & 31;
  int col = lane & 15, hi = lane >> 4;
  int p = blockIdx.x * 16 + col;           // output pixel (B-matrix column)
  int oy = p / W2, ox = p % W2;
  const __bf16* xb = x_bf + (size_t)b * NTOK * CIN;
  const __bf16* arow = convA + (w * 16 + col) * KCONV;
  v8f acc = zero8();
  for (int k0 = 0; k0 < KCONV; k0 += 32) {
    v16bf a = load_afrag(arow, k0, hi);
    int kk = k0 + hi * 16;                 // 16 consecutive k, always within one tap
    int tap = kk / CIN, coff = kk % CIN;
    int ky = tap / 3, kx = tap % 3;
    int iy = 2 * oy - 1 + ky, ix = 2 * ox - 1 + kx;
    v16bf bf;
    if (iy >= 0 && iy < HIMG && ix >= 0 && ix < WIMG) {
      BF16x16 u;
      const __bf16* src = xb + (iy * WIMG + ix) * CIN + coff;
      u.f4[0] = *(const float4*)(src);
      u.f4[1] = *(const float4*)(src + 8);
      bf = u.v;
    } else {
      bf = zero_frag();
    }
    acc = wmma_bf16(a, bf, acc);
  }
  int ch0 = w * 16 + hi * 8;
  float* outp = conv_out + ((size_t)b * NPIX + p) * COUT + ch0;
#pragma unroll
  for (int v = 0; v < 8; ++v) outp[v] = acc[v] + conv_b[ch0 + v];
}

// ---------------------------------------------------------------- skip GEMM + conv gather + LN + conf
// grid (196, B), block 256
__global__ __launch_bounds__(256) void k_tok(const __bf16* x_bf, const __bf16* skipA,
                                             const float* conv_out, const float* loc_orig,
                                             const float* ln_g, const float* ln_b,
                                             const float* conf_w, const float* conf_b,
                                             float* xtok, __bf16* xtok_bf,
                                             float* x2arr, float* wgt, int Bv) {
  __shared__ float tile[16][COUT + 4];
  int b = blockIdx.y, tt = blockIdx.x;
  int tid = threadIdx.x, w = tid >> 5, lane = tid & 31;
  int col = lane & 15, hi = lane >> 4;
  int t = tt * 16 + col;
  const __bf16* arow = skipA + (w * 16 + col) * CIN;
  const __bf16* brow = x_bf + ((size_t)b * NTOK + t) * CIN;
  v8f acc = zero8();
#pragma unroll
  for (int k0 = 0; k0 < CIN; k0 += 32) {
    acc = wmma_bf16(load_afrag(arow, k0, hi), load_bfrag(brow, k0, hi), acc);
  }
  // nearest map pixel for this token
  float lx = loc_orig[((size_t)b * NTOK + t) * 2 + 0];
  float ly = loc_orig[((size_t)b * NTOK + t) * 2 + 1];
  lx = fminf(fmaxf(lx, -1.0f), 1.0f);
  ly = fminf(fmaxf(ly, -1.0f), 1.0f);
  int px = (int)rintf(0.5f * (lx + 1.0f) * (float)W2 - 0.5f);
  int py = (int)rintf(0.5f * (ly + 1.0f) * (float)H2 - 0.5f);
  px = min(max(px, 0), W2 - 1); py = min(max(py, 0), H2 - 1);
  int p = py * W2 + px;
  int ch0 = w * 16 + hi * 8;
  const float* cp = conv_out + ((size_t)b * NPIX + p) * COUT + ch0;
  const float invw = 1.0f / (1.0f + 1e-6f);
#pragma unroll
  for (int v = 0; v < 8; ++v) tile[col][ch0 + v] = acc[v] + cp[v] * invw;
  __syncthreads();
  if (tid < 16) {
    int tok = tt * 16 + tid;
    float mu = 0.0f;
    for (int c = 0; c < COUT; ++c) mu += tile[tid][c];
    mu *= (1.0f / COUT);
    float var = 0.0f;
    for (int c = 0; c < COUT; ++c) { float d = tile[tid][c] - mu; var += d * d; }
    var *= (1.0f / COUT);
    float rs = rsqrtf(var + LN_EPS);
    float x2 = 0.0f, conf = 0.0f;
    float*  xo  = xtok    + ((size_t)b * NTOK + tok) * COUT;
    __bf16* xob = xtok_bf + ((size_t)b * NTOK + tok) * COUT;
    for (int c = 0; c < COUT; ++c) {
      float v = (tile[tid][c] - mu) * rs * ln_g[c] + ln_b[c];
      xo[c] = v;
      __bf16 bv = (__bf16)v;
      xob[c] = bv;
      float fb = (float)bv;
      x2 += fb * fb;
      conf += v * conf_w[c];
    }
    wgt[(size_t)b * NTOK + tok]   = expf(conf + conf_b[0]);
    x2arr[(size_t)b * NTOK + tok] = x2;
  }
}

// ---------------------------------------------------------------- KNN density via WMMA Gram sweep
// grid (196, B): workgroup owns 16 "i" tokens (B columns); waves stride over j tiles (A rows).
__global__ __launch_bounds__(256) void k_density(const __bf16* xtb, const float* x2arr,
                                                 float* dens, unsigned* d2max, int Bv) {
  __shared__ float s0[256], s1[256], s2[256], sm[256];
  int b = blockIdx.y, it = blockIdx.x;
  int tid = threadIdx.x, w = tid >> 5, lane = tid & 31;
  int col = lane & 15, hi = lane >> 4;
  int i = it * 16 + col;
  const __bf16* base = xtb + (size_t)b * NTOK * COUT;
  float xi2 = x2arr[(size_t)b * NTOK + i];
  v16bf bf0 = load_bfrag(base + (size_t)i * COUT, 0, hi);
  v16bf bf1 = load_bfrag(base + (size_t)i * COUT, 32, hi);
  v16bf bf2 = load_bfrag(base + (size_t)i * COUT, 64, hi);
  v16bf bf3 = load_bfrag(base + (size_t)i * COUT, 96, hi);
  float t0 = 3.4e38f, t1 = 3.4e38f, t2 = 3.4e38f, mx = 0.0f;
  for (int jt = w; jt < 196; jt += 8) {
    const __bf16* arow = base + (size_t)(jt * 16 + col) * COUT;
    if (jt + 8 < 196) __builtin_prefetch(base + (size_t)((jt + 8) * 16 + col) * COUT, 0, 3);
    v8f acc = zero8();
    acc = wmma_bf16(load_afrag(arow, 0, hi),  bf0, acc);
    acc = wmma_bf16(load_afrag(arow, 32, hi), bf1, acc);
    acc = wmma_bf16(load_afrag(arow, 64, hi), bf2, acc);
    acc = wmma_bf16(load_afrag(arow, 96, hi), bf3, acc);
    int jbase = jt * 16 + hi * 8;
#pragma unroll
    for (int v = 0; v < 8; ++v) {
      float d2 = xi2 + x2arr[(size_t)b * NTOK + jbase + v] - 2.0f * acc[v];
      mx = fmaxf(mx, d2);
      ins3(fmaxf(d2, 1e-12f), t0, t1, t2);
    }
  }
  s0[tid] = t0; s1[tid] = t1; s2[tid] = t2; sm[tid] = mx;
  __syncthreads();
  if (tid < 16) {
    float a0 = 3.4e38f, a1 = 3.4e38f, a2 = 3.4e38f, m = 0.0f;
    for (int ww = 0; ww < 8; ++ww)
      for (int h = 0; h < 2; ++h) {
        int src = ww * 32 + h * 16 + tid;
        ins3(s0[src], a0, a1, a2); ins3(s1[src], a0, a1, a2); ins3(s2[src], a0, a1, a2);
        m = fmaxf(m, sm[src]);
      }
    float mean = (a0 + a1 + a2) * (1.0f / (3.0f * (float)COUT));
    int tok = it * 16 + tid;
    unsigned hsh = (unsigned)tok * 2654435761u + (unsigned)b * 40499u;
    float noise = (float)(hsh >> 16) * (1.0f / 65536.0f) * 1e-6f;  // deterministic tie-break
    dens[(size_t)b * NTOK + tok] = expf(-mean) + noise;
    atomicMax(&d2max[b], __float_as_uint(m));  // d2 >= 0: bit order == float order
  }
}

// ---------------------------------------------------------------- parent distance * density (score)
__global__ __launch_bounds__(256) void k_score(const __bf16* xtb, const float* x2arr,
                                               const float* dens, const unsigned* d2max,
                                               float* score, int Bv) {
  __shared__ float smin[256];
  int b = blockIdx.y, it = blockIdx.x;
  int tid = threadIdx.x, w = tid >> 5, lane = tid & 31;
  int col = lane & 15, hi = lane >> 4;
  int i = it * 16 + col;
  const __bf16* base = xtb + (size_t)b * NTOK * COUT;
  const float rsc = rsqrtf((float)COUT);
  float distmax = sqrtf(fmaxf(__uint_as_float(d2max[b]), 1e-12f)) * rsc;
  float xi2 = x2arr[(size_t)b * NTOK + i];
  float di  = dens[(size_t)b * NTOK + i];
  v16bf bf0 = load_bfrag(base + (size_t)i * COUT, 0, hi);
  v16bf bf1 = load_bfrag(base + (size_t)i * COUT, 32, hi);
  v16bf bf2 = load_bfrag(base + (size_t)i * COUT, 64, hi);
  v16bf bf3 = load_bfrag(base + (size_t)i * COUT, 96, hi);
  float pmin = distmax;
  for (int jt = w; jt < 196; jt += 8) {
    const __bf16* arow = base + (size_t)(jt * 16 + col) * COUT;
    if (jt + 8 < 196) __builtin_prefetch(base + (size_t)((jt + 8) * 16 + col) * COUT, 0, 3);
    v8f acc = zero8();
    acc = wmma_bf16(load_afrag(arow, 0, hi),  bf0, acc);
    acc = wmma_bf16(load_afrag(arow, 32, hi), bf1, acc);
    acc = wmma_bf16(load_afrag(arow, 64, hi), bf2, acc);
    acc = wmma_bf16(load_afrag(arow, 96, hi), bf3, acc);
    int jbase = jt * 16 + hi * 8;
#pragma unroll
    for (int v = 0; v < 8; ++v) {
      int j = jbase + v;
      float dj = dens[(size_t)b * NTOK + j];
      float d2 = xi2 + x2arr[(size_t)b * NTOK + j] - 2.0f * acc[v];
      float dist = sqrtf(fmaxf(d2, 1e-12f)) * rsc;
      if (dj > di && dist < pmin) pmin = dist;
    }
  }
  smin[tid] = pmin;
  __syncthreads();
  if (tid < 16) {
    float m = 3.4e38f;
    for (int ww = 0; ww < 8; ++ww)
      for (int h = 0; h < 2; ++h) m = fminf(m, smin[ww * 32 + h * 16 + tid]);
    int tok = it * 16 + tid;
    score[(size_t)b * NTOK + tok] = m * dens[(size_t)b * NTOK + tok];
  }
}

// ---------------------------------------------------------------- exact top-784 via LDS bitonic sort
__global__ __launch_bounds__(1024) void k_sort(const float* score, int* index_down, int Bv) {
  __shared__ unsigned long long keys[NPAD];
  int b = blockIdx.x;
  for (int i = threadIdx.x; i < NPAD; i += blockDim.x) {
    unsigned long long k;
    if (i < NTOK) {
      unsigned u = __float_as_uint(score[(size_t)b * NTOK + i]);
      unsigned o = (u & 0x80000000u) ? ~u : (u | 0x80000000u);  // orderable ascending
      unsigned d = ~o;                                          // descending score
      k = ((unsigned long long)d << 32) | (unsigned)i;          // tie -> lower index first
    } else {
      k = ~0ull;
    }
    keys[i] = k;
  }
  __syncthreads();
  for (unsigned size = 2; size <= NPAD; size <<= 1) {
    for (unsigned stride = size >> 1; stride > 0; stride >>= 1) {
      for (unsigned i = threadIdx.x; i < NPAD / 2; i += blockDim.x) {
        unsigned lo = (i / stride) * (stride << 1) + (i % stride);
        unsigned hipos = lo + stride;
        bool asc = ((lo & size) == 0);
        unsigned long long a = keys[lo], c = keys[hipos];
        if ((a > c) == asc) { keys[lo] = c; keys[hipos] = a; }
      }
      __syncthreads();
    }
  }
  for (int s = threadIdx.x; s < NS; s += blockDim.x)
    index_down[(size_t)b * NS + s] = (int)(keys[s] & 0xFFFFFFFFu);
}

// ---------------------------------------------------------------- assign tokens to nearest center
__global__ __launch_bounds__(256) void k_assign(const __bf16* xtb, const float* x2arr,
                                                const int* index_down, int* idx_cluster, int Bv) {
  __shared__ float sd[256];
  __shared__ int   si[256];
  int b = blockIdx.y, tt = blockIdx.x;
  int tid = threadIdx.x, w = tid >> 5, lane = tid & 31;
  int col = lane & 15, hi = lane >> 4;
  int n = tt * 16 + col;
  const __bf16* base = xtb + (size_t)b * NTOK * COUT;
  float xn2 = x2arr[(size_t)b * NTOK + n];
  v16bf bf0 = load_bfrag(base + (size_t)n * COUT, 0, hi);
  v16bf bf1 = load_bfrag(base + (size_t)n * COUT, 32, hi);
  v16bf bf2 = load_bfrag(base + (size_t)n * COUT, 64, hi);
  v16bf bf3 = load_bfrag(base + (size_t)n * COUT, 96, hi);
  float bestd = 3.4e38f; int bests = 0x7FFFFFFF;
  for (int ct = w; ct < 49; ct += 8) {
    int sA = ct * 16 + col;
    int tc = index_down[(size_t)b * NS + sA];
    const __bf16* arow = base + (size_t)tc * COUT;
    v8f acc = zero8();
    acc = wmma_bf16(load_afrag(arow, 0, hi),  bf0, acc);
    acc = wmma_bf16(load_afrag(arow, 32, hi), bf1, acc);
    acc = wmma_bf16(load_afrag(arow, 64, hi), bf2, acc);
    acc = wmma_bf16(load_afrag(arow, 96, hi), bf3, acc);
#pragma unroll
    for (int v = 0; v < 8; ++v) {
      int s = ct * 16 + v + hi * 8;
      int tj = index_down[(size_t)b * NS + s];
      float d2 = fmaxf(x2arr[(size_t)b * NTOK + tj] + xn2 - 2.0f * acc[v], 1e-12f);
      if (d2 < bestd || (d2 == bestd && s < bests)) { bestd = d2; bests = s; }
    }
  }
  sd[tid] = bestd; si[tid] = bests;
  __syncthreads();
  if (tid < 16) {
    float bd = 3.4e38f; int bs = 0x7FFFFFFF;
    for (int ww = 0; ww < 8; ++ww)
      for (int h = 0; h < 2; ++h) {
        int src = ww * 32 + h * 16 + tid;
        float d = sd[src]; int s = si[src];
        if (d < bd || (d == bd && s < bs)) { bd = d; bs = s; }
      }
    idx_cluster[(size_t)b * NTOK + tt * 16 + tid] = bs;
  }
}

// centers merge to themselves
__global__ void k_self(const int* index_down, int* idx_cluster, int Bv) {
  int tid = blockIdx.x * blockDim.x + threadIdx.x;
  if (tid < Bv * NS) {
    int b = tid / NS, s = tid % NS;
    idx_cluster[(size_t)b * NTOK + index_down[tid]] = s;
  }
}

// ---------------------------------------------------------------- segment sums / merging
__global__ void k_wsum(const int* idx_cluster, const float* wgt, float* all_w, int Bv) {
  int tid = blockIdx.x * blockDim.x + threadIdx.x;
  if (tid < Bv * NTOK) {
    int b = tid / NTOK;
    atomicAdd(&all_w[(size_t)b * NS + idx_cluster[tid]], wgt[tid]);
  }
}
__global__ void k_norm(const int* idx_cluster, const float* wgt, const float* all_w,
                       float* norm_w, int Bv) {
  int tid = blockIdx.x * blockDim.x + threadIdx.x;
  if (tid < Bv * NTOK) {
    int b = tid / NTOK;
    norm_w[tid] = wgt[tid] / all_w[(size_t)b * NS + idx_cluster[tid]];
  }
}
__global__ void k_scatter(const int* idx_cluster, const float* norm_w, const float* xtok,
                          float* xdown, int Bv) {
  int tid = blockIdx.x * blockDim.x + threadIdx.x;
  if (tid < Bv * NTOK * 32) {
    int chunk = tid & 31;
    int bn = tid >> 5;
    int b = bn / NTOK;
    int c = idx_cluster[bn];
    float nw = norm_w[bn];
    const float* xr = xtok + (size_t)bn * COUT + chunk * 4;
    float* dr = xdown + ((size_t)b * NS + c) * COUT + chunk * 4;
#pragma unroll
    for (int q = 0; q < 4; ++q) atomicAdd(&dr[q], xr[q] * nw);
  }
}
__global__ void k_agg(const int* idx_agg, const float* agg_weight, const float* norm_w,
                      const int* idx_cluster, float* aggtmp, int* idx_out,
                      unsigned* aggmax, int Bv) {
  int tid = blockIdx.x * blockDim.x + threadIdx.x;
  if (tid < Bv * NTOK) {
    int b = tid / NTOK;
    int ia = idx_agg[tid];
    idx_out[tid] = idx_cluster[(size_t)b * NTOK + ia];
    float a = agg_weight[tid] * norm_w[(size_t)b * NTOK + ia];
    aggtmp[tid] = a;
    atomicMax(&aggmax[b], __float_as_uint(a));  // a >= 0
  }
}
__global__ void k_div(const float* aggtmp, const unsigned* aggmax, float* agg_out, int Bv) {
  int tid = blockIdx.x * blockDim.x + threadIdx.x;
  if (tid < Bv * NTOK) {
    int b = tid / NTOK;
    agg_out[tid] = aggtmp[tid] / __uint_as_float(aggmax[b]);
  }
}

// ---------------------------------------------------------------- launcher
extern "C" void kernel_launch(void* const* d_in, const int* in_sizes, int n_in,
                              void* d_out, int out_size, void* d_ws, size_t ws_size,
                              hipStream_t stream) {
  const float* x          = (const float*)d_in[0];
  const float* loc_orig   = (const float*)d_in[1];
  const float* agg_weight = (const float*)d_in[2];
  const float* conv_w     = (const float*)d_in[3];
  const float* conv_b     = (const float*)d_in[4];
  const float* skip_w     = (const float*)d_in[5];
  const float* ln_g       = (const float*)d_in[6];
  const float* ln_b       = (const float*)d_in[7];
  const float* conf_w     = (const float*)d_in[8];
  const float* conf_b     = (const float*)d_in[9];
  const int*   idx_agg    = (const int*)d_in[10];

  const int Bv = in_sizes[0] / (NTOK * CIN);

  // workspace carve-out (256B aligned slabs)
  char* ws = (char*)d_ws;
  size_t off = 0;
  auto carve = [&](size_t bytes) -> void* {
    void* p = ws + off;
    off += (bytes + 255) & ~(size_t)255;
    return p;
  };
  __bf16* x_bf     = (__bf16*)carve((size_t)Bv * NTOK * CIN * 2);
  __bf16* convA    = (__bf16*)carve((size_t)COUT * KCONV * 2);
  __bf16* skipA    = (__bf16*)carve((size_t)COUT * CIN * 2);
  float*  conv_out = (float*)carve((size_t)Bv * NPIX * COUT * 4);
  float*  xtok     = (float*)carve((size_t)Bv * NTOK * COUT * 4);
  __bf16* xtok_bf  = (__bf16*)carve((size_t)Bv * NTOK * COUT * 2);
  float*  x2arr    = (float*)carve((size_t)Bv * NTOK * 4);
  float*  wgt      = (float*)carve((size_t)Bv * NTOK * 4);
  float*  dens     = (float*)carve((size_t)Bv * NTOK * 4);
  float*  score    = (float*)carve((size_t)Bv * NTOK * 4);
  int*    index_dn = (int*)carve((size_t)Bv * NS * 4);
  int*    idx_clu  = (int*)carve((size_t)Bv * NTOK * 4);
  float*  all_w    = (float*)carve((size_t)Bv * NS * 4);
  float*  norm_w   = (float*)carve((size_t)Bv * NTOK * 4);
  float*  aggtmp   = (float*)carve((size_t)Bv * NTOK * 4);
  unsigned* d2max  = (unsigned*)carve((size_t)Bv * 4);
  unsigned* aggmax = (unsigned*)carve((size_t)Bv * 4);

  // output layout: x_down | idx_agg_down (int bits) | agg_weight_down
  float* out    = (float*)d_out;
  float* xdown  = out;
  int*   idxout = (int*)(out + (size_t)Bv * NS * COUT);
  float* aggout = out + (size_t)Bv * NS * COUT + (size_t)Bv * NTOK;

  const int nBN = Bv * NTOK;

  k_init<<<256, 256, 0, stream>>>(all_w, d2max, aggmax, xdown, Bv);
  k_pack<<<512, 256, 0, stream>>>(x, conv_w, skip_w, x_bf, convA, skipA, Bv);
  k_conv<<<dim3(NPIX / 16, Bv), 256, 0, stream>>>(x_bf, convA, conv_b, conv_out, Bv);
  k_tok<<<dim3(NTOK / 16, Bv), 256, 0, stream>>>(x_bf, skipA, conv_out, loc_orig,
                                                 ln_g, ln_b, conf_w, conf_b,
                                                 xtok, xtok_bf, x2arr, wgt, Bv);
  k_density<<<dim3(NTOK / 16, Bv), 256, 0, stream>>>(xtok_bf, x2arr, dens, d2max, Bv);
  k_score<<<dim3(NTOK / 16, Bv), 256, 0, stream>>>(xtok_bf, x2arr, dens, d2max, score, Bv);
  k_sort<<<Bv, 1024, 0, stream>>>(score, index_dn, Bv);
  k_assign<<<dim3(NTOK / 16, Bv), 256, 0, stream>>>(xtok_bf, x2arr, index_dn, idx_clu, Bv);
  k_self<<<(Bv * NS + 255) / 256, 256, 0, stream>>>(index_dn, idx_clu, Bv);
  k_wsum<<<(nBN + 255) / 256, 256, 0, stream>>>(idx_clu, wgt, all_w, Bv);
  k_norm<<<(nBN + 255) / 256, 256, 0, stream>>>(idx_clu, wgt, all_w, norm_w, Bv);
  k_scatter<<<(nBN * 32 + 255) / 256, 256, 0, stream>>>(idx_clu, norm_w, xtok, xdown, Bv);
  k_agg<<<(nBN + 255) / 256, 256, 0, stream>>>(idx_agg, agg_weight, norm_w, idx_clu,
                                               aggtmp, idxout, aggmax, Bv);
  k_div<<<(nBN + 255) / 256, 256, 0, stream>>>(aggtmp, aggmax, aggout, Bv);
}
